// CaptionEmbedding_72559177499102
// MI455X (gfx1250) — matrix operand order repaired and checked by reading, not compiled, not run on hardware
//
#include <hip/hip_runtime.h>
#include <hip/hip_bf16.h>
#include <math.h>

// Problem dims (fixed by the reference)
constexpr int BB = 2048, TT = 20, CD = 512, HD = 1024, VD = 2048, QD = 1024;

typedef __attribute__((ext_vector_type(16))) __bf16 v16bf;
typedef __attribute__((ext_vector_type(8)))  __bf16 v8bf;
typedef __attribute__((ext_vector_type(8)))  float  v8f;

__device__ __forceinline__ float sigmoid_f(float x) { return 1.0f / (1.0f + expf(-x)); }
__device__ __forceinline__ float leaky_f(float x)   { return x > 0.0f ? x : 0.01f * x; }

__device__ __forceinline__ v16bf pack16(v8bf lo, v8bf hi) {
    v16bf f;
#pragma unroll
    for (int e = 0; e < 8; ++e) { f[e] = lo[e]; f[e + 8] = hi[e]; }
    return f;
}

// ---------------------------------------------------------------------------
// fp32 -> bf16 conversion (contiguous)
// ---------------------------------------------------------------------------
__global__ void cvt_f32_bf16(const float* __restrict__ s, __bf16* __restrict__ d, int n) {
    int i = blockIdx.x * blockDim.x + threadIdx.x;
    if (i < n) d[i] = (__bf16)s[i];
}

// Convert caption[:, t, :] -> xt_bf [BB, CD]
__global__ void cvt_caption_slice(const float* __restrict__ caption,
                                  __bf16* __restrict__ xt, int t) {
    int idx = blockIdx.x * blockDim.x + threadIdx.x;
    if (idx >= BB * CD) return;
    int b = idx >> 9, c = idx & 511;
    xt[idx] = (__bf16)caption[((size_t)b * TT + t) * CD + c];
}

// ---------------------------------------------------------------------------
// init: h1 = 0 (f32+bf16), h2 = 0 (f32+bf16), out = -inf-ish
// ---------------------------------------------------------------------------
__global__ void init_state(float* __restrict__ h1, __bf16* __restrict__ h1b,
                           float* __restrict__ h2, __bf16* __restrict__ h2b,
                           float* __restrict__ outp) {
    int i = blockIdx.x * blockDim.x + threadIdx.x;
    if (i < BB * HD) { h2[i] = 0.0f; h2b[i] = (__bf16)0.0f; outp[i] = -3.0e38f; }
    if (i < BB * CD) { h1[i] = 0.0f; h1b[i] = (__bf16)0.0f; }
}

// ---------------------------------------------------------------------------
// Double-buffered, register-prefetch pipelined WMMA GEMM:
//   C[M,N] = A[M,K](bf16, row stride lda) x W[N,K](bf16, row-major)^T
// flags: 0 = plain store, 1 = leaky_relu on store
// Block: 256 threads = 8 waves. Block tile 128(M) x 128(N), K-slab 64.
// Wave tile 32(M) x 64(N) = 2x4 grid of 16x16 accumulators, 16 WMMAs/slab.
// LDS: 2 x (16KB A + 16KB B) slabs (+pad) = 72KB -> 4 blocks/WGP.
// One barrier per slab; next slab's global loads overlap current compute.
// Requires M % 128 == 0, N % 128 == 0, K % 64 == 0 (true for all call sites).
// ---------------------------------------------------------------------------
__global__ __launch_bounds__(256)
void gemm_bf16_wmma(const __bf16* __restrict__ A, int lda,
                    const __bf16* __restrict__ W,
                    float* __restrict__ C, int N, int K, int flags) {
    __shared__ __bf16 As[2][128][72];   // 64 K-cols + 8 pad
    __shared__ __bf16 Bs[2][128][72];

    const int tid   = threadIdx.x;
    const int lane  = tid & 31;
    const int wave  = tid >> 5;
    const int waveM = wave & 3;    // 4 waves along M, 32 rows each
    const int waveN = wave >> 2;   // 2 waves along N, 64 cols each
    const int mBlock = blockIdx.y * 128;
    const int nBlock = blockIdx.x * 128;
    const int m    = lane & 15;
    const int half = lane >> 4;

    // Per-thread staging coordinates: 4 chunks of 8 bf16 per array.
    int srow[4], scol[4];
#pragma unroll
    for (int i = 0; i < 4; ++i) {
        int cg  = i * 256 + tid;       // 1024 chunks of 8 bf16
        srow[i] = cg >> 3;
        scol[i] = (cg & 7) * 8;
    }

    v8f acc[2][4];
#pragma unroll
    for (int s = 0; s < 2; ++s)
#pragma unroll
        for (int j = 0; j < 4; ++j)
#pragma unroll
            for (int e = 0; e < 8; ++e) acc[s][j][e] = 0.0f;

    v8bf ra[4], rb[4];

    // ---- prologue: fetch slab 0 into registers, stage into LDS buffer 0
#pragma unroll
    for (int i = 0; i < 4; ++i) {
        ra[i] = *(const v8bf*)(A + (size_t)(mBlock + srow[i]) * lda + scol[i]);
        rb[i] = *(const v8bf*)(W + (size_t)(nBlock + srow[i]) * K   + scol[i]);
    }
#pragma unroll
    for (int i = 0; i < 4; ++i) {
        *(v8bf*)&As[0][srow[i]][scol[i]] = ra[i];
        *(v8bf*)&Bs[0][srow[i]][scol[i]] = rb[i];
    }
    __syncthreads();

    int cur = 0;
    for (int k0 = 0; k0 < K; k0 += 64) {
        const bool more = (k0 + 64) < K;
        // ---- prefetch next slab into registers (loads overlap the WMMAs below)
        if (more) {
#pragma unroll
            for (int i = 0; i < 4; ++i) {
                ra[i] = *(const v8bf*)(A + (size_t)(mBlock + srow[i]) * lda + k0 + 64 + scol[i]);
                rb[i] = *(const v8bf*)(W + (size_t)(nBlock + srow[i]) * K   + k0 + 64 + scol[i]);
            }
        }

        // ---- compute current slab: two K=32 sub-steps, 16 WMMAs per wave
#pragma unroll
        for (int ks = 0; ks < 64; ks += 32) {
            v16bf af[2];
#pragma unroll
            for (int s = 0; s < 2; ++s) {
                const __bf16* pr = &As[cur][waveM * 32 + s * 16 + m][ks + 8 * half];
                af[s] = pack16(*(const v8bf*)pr, *(const v8bf*)(pr + 16));
            }
            v16bf bfr[4];
#pragma unroll
            for (int j = 0; j < 4; ++j) {
                const __bf16* pr = &Bs[cur][waveN * 64 + j * 16 + m][ks + 8 * half];
                bfr[j] = pack16(*(const v8bf*)pr, *(const v8bf*)(pr + 16));
            }
#pragma unroll
            for (int s = 0; s < 2; ++s)
#pragma unroll
                for (int j = 0; j < 4; ++j)
                    acc[s][j] = __builtin_amdgcn_wmma_f32_16x16x32_bf16(
                        false, af[s], false, bfr[j], (short)0, acc[s][j], false, false);
        }

        // ---- stage next slab into the other LDS buffer; one barrier per slab
        if (more) {
            int nxt = cur ^ 1;
#pragma unroll
            for (int i = 0; i < 4; ++i) {
                *(v8bf*)&As[nxt][srow[i]][scol[i]] = ra[i];
                *(v8bf*)&Bs[nxt][srow[i]][scol[i]] = rb[i];
            }
            __syncthreads();
            cur = nxt;
        }
    }

    // ---- epilogue
#pragma unroll
    for (int s = 0; s < 2; ++s) {
#pragma unroll
        for (int j = 0; j < 4; ++j) {
            const int col = nBlock + waveN * 64 + j * 16 + m;
#pragma unroll
            for (int r = 0; r < 8; ++r) {
                const int row = mBlock + waveM * 32 + s * 16 + 8 * half + r;
                float v = acc[s][j][r];
                if (flags == 1) v = leaky_f(v);
                C[(size_t)row * N + col] = v;
            }
        }
    }
}

// ---------------------------------------------------------------------------
// GRU1 gates + attention + alphas.  One thread per (b, c), c < CD.
// Writes fp32 master h1, bf16 shadow h1b, bf16 att (GEMM operand), alphas.
// ---------------------------------------------------------------------------
__global__ void gru1_att_kernel(const float* __restrict__ gi, const float* __restrict__ gh,
                                const float* __restrict__ b_ih, const float* __restrict__ b_hh,
                                float* __restrict__ h1, __bf16* __restrict__ h1b,
                                const float* __restrict__ fv, const float* __restrict__ fq,
                                const float* __restrict__ caption,
                                const int* __restrict__ cap_len,
                                __bf16* __restrict__ attb, float* __restrict__ alphas, int t) {
    int idx = blockIdx.x * blockDim.x + threadIdx.x;
    if (idx >= BB * CD) return;
    const int b = idx >> 9;     // CD = 512
    const int c = idx & 511;
    const size_t g0 = (size_t)b * 3 * CD;
    float ir  = gi[g0 + c]           + b_ih[c];
    float iz  = gi[g0 + CD + c]      + b_ih[CD + c];
    float in_ = gi[g0 + 2 * CD + c]  + b_ih[2 * CD + c];
    float hr  = gh[g0 + c]           + b_hh[c];
    float hz  = gh[g0 + CD + c]      + b_hh[CD + c];
    float hn  = gh[g0 + 2 * CD + c]  + b_hh[2 * CD + c];
    float r  = sigmoid_f(ir + hr);
    float z  = sigmoid_f(iz + hz);
    float nn = tanhf(in_ + r * hn);
    float hold = h1[idx];
    float hnew = (1.0f - z) * nn + z * hold;
    bool active = t < cap_len[b];
    float hv = active ? hnew : hold;
    h1[idx]  = hv;
    h1b[idx] = (__bf16)hv;
    float x = caption[((size_t)b * TT + t) * CD + c];
    float a = sigmoid_f(hv * fv[idx] + hv * fq[idx]) * x;
    attb[idx] = (__bf16)a;
    alphas[((size_t)b * TT + t) * CD + c] = active ? a : 0.0f;
}

// ---------------------------------------------------------------------------
// GRU2 gates.  One thread per (b, h), h < HD.
// ---------------------------------------------------------------------------
__global__ void gru2_kernel(const float* __restrict__ gi, const float* __restrict__ gh,
                            const float* __restrict__ b_ih, const float* __restrict__ b_hh,
                            float* __restrict__ h2, __bf16* __restrict__ h2b,
                            const int* __restrict__ cap_len, int t) {
    int idx = blockIdx.x * blockDim.x + threadIdx.x;
    if (idx >= BB * HD) return;
    const int b = idx >> 10;    // HD = 1024
    const int c = idx & 1023;
    const size_t g0 = (size_t)b * 3 * HD;
    float ir  = gi[g0 + c]           + b_ih[c];
    float iz  = gi[g0 + HD + c]      + b_ih[HD + c];
    float in_ = gi[g0 + 2 * HD + c]  + b_ih[2 * HD + c];
    float hr  = gh[g0 + c]           + b_hh[c];
    float hz  = gh[g0 + HD + c]      + b_hh[HD + c];
    float hn  = gh[g0 + 2 * HD + c]  + b_hh[2 * HD + c];
    float r  = sigmoid_f(ir + hr);
    float z  = sigmoid_f(iz + hz);
    float nn = tanhf(in_ + r * hn);
    float hold = h2[idx];
    float hnew = (1.0f - z) * nn + z * hold;
    bool active = t < cap_len[b];
    float hv = active ? hnew : hold;
    h2[idx]  = hv;
    h2b[idx] = (__bf16)hv;
}

// ---------------------------------------------------------------------------
// Masked leaky + running max-pool into output.
// ---------------------------------------------------------------------------
__global__ void maxpool_kernel(const float* __restrict__ o, const int* __restrict__ cap_len,
                               float* __restrict__ outp, int t) {
    int idx = blockIdx.x * blockDim.x + threadIdx.x;
    if (idx >= BB * HD) return;
    const int b = idx >> 10;
    bool active = t < cap_len[b];
    float v = active ? leaky_f(o[idx]) : 0.0f;
    outp[idx] = fmaxf(outp[idx], v);
}

// ---------------------------------------------------------------------------
// Workspace layout
// ---------------------------------------------------------------------------
static constexpr size_t alignup(size_t x) { return (x + 255) & ~(size_t)255; }

constexpr size_t OFF_WV   = 0;
constexpr size_t OFF_WQ   = alignup(OFF_WV   + (size_t)CD * VD * 2);
constexpr size_t OFF_WIH1 = alignup(OFF_WQ   + (size_t)CD * QD * 2);
constexpr size_t OFF_WHH1 = alignup(OFF_WIH1 + (size_t)3 * CD * CD * 2);
constexpr size_t OFF_WIH2 = alignup(OFF_WHH1 + (size_t)3 * CD * CD * 2);
constexpr size_t OFF_WHH2 = alignup(OFF_WIH2 + (size_t)3 * HD * CD * 2);
constexpr size_t OFF_WFC  = alignup(OFF_WHH2 + (size_t)3 * HD * HD * 2);
constexpr size_t OFF_VB   = alignup(OFF_WFC  + (size_t)HD * HD * 2);
constexpr size_t OFF_QB   = alignup(OFF_VB   + (size_t)BB * VD * 2);
constexpr size_t OFF_FV   = alignup(OFF_QB   + (size_t)BB * QD * 2);
constexpr size_t OFF_FQ   = alignup(OFF_FV   + (size_t)BB * CD * 4);
constexpr size_t OFF_XT   = alignup(OFF_FQ   + (size_t)BB * CD * 4);
constexpr size_t OFF_GI1  = alignup(OFF_XT   + (size_t)BB * CD * 2);
constexpr size_t OFF_GH1  = alignup(OFF_GI1  + (size_t)BB * 3 * CD * 4);
constexpr size_t OFF_H1   = alignup(OFF_GH1  + (size_t)BB * 3 * CD * 4);
constexpr size_t OFF_H1B  = alignup(OFF_H1   + (size_t)BB * CD * 4);
constexpr size_t OFF_ATTB = alignup(OFF_H1B  + (size_t)BB * CD * 2);
constexpr size_t OFF_GI2  = alignup(OFF_ATTB + (size_t)BB * CD * 2);
constexpr size_t OFF_GH2  = alignup(OFF_GI2  + (size_t)BB * 3 * HD * 4);
constexpr size_t OFF_H2   = alignup(OFF_GH2  + (size_t)BB * 3 * HD * 4);
constexpr size_t OFF_H2B  = alignup(OFF_H2   + (size_t)BB * HD * 4);

extern "C" void kernel_launch(void* const* d_in, const int* in_sizes, int n_in,
                              void* d_out, int out_size, void* d_ws, size_t ws_size,
                              hipStream_t stream) {
    const float* v       = (const float*)d_in[0];
    const float* q       = (const float*)d_in[1];
    const float* caption = (const float*)d_in[2];
    const int*   cap_len = (const int*)d_in[3];
    const float* w_ih1   = (const float*)d_in[4];
    const float* w_hh1   = (const float*)d_in[5];
    const float* b_ih1   = (const float*)d_in[6];
    const float* b_hh1   = (const float*)d_in[7];
    const float* w_ih2   = (const float*)d_in[8];
    const float* w_hh2   = (const float*)d_in[9];
    const float* b_ih2   = (const float*)d_in[10];
    const float* b_hh2   = (const float*)d_in[11];
    const float* Wv      = (const float*)d_in[12];
    const float* Wq      = (const float*)d_in[13];
    const float* Wfc     = (const float*)d_in[14];

    char* ws = (char*)d_ws;
    __bf16* wv_b   = (__bf16*)(ws + OFF_WV);
    __bf16* wq_b   = (__bf16*)(ws + OFF_WQ);
    __bf16* wih1_b = (__bf16*)(ws + OFF_WIH1);
    __bf16* whh1_b = (__bf16*)(ws + OFF_WHH1);
    __bf16* wih2_b = (__bf16*)(ws + OFF_WIH2);
    __bf16* whh2_b = (__bf16*)(ws + OFF_WHH2);
    __bf16* wfc_b  = (__bf16*)(ws + OFF_WFC);
    __bf16* v_b    = (__bf16*)(ws + OFF_VB);
    __bf16* q_b    = (__bf16*)(ws + OFF_QB);
    float*  fv     = (float*)(ws + OFF_FV);
    float*  fq     = (float*)(ws + OFF_FQ);
    __bf16* xt_b   = (__bf16*)(ws + OFF_XT);
    float*  gi1    = (float*)(ws + OFF_GI1);
    float*  gh1    = (float*)(ws + OFF_GH1);
    float*  h1     = (float*)(ws + OFF_H1);
    __bf16* h1_b   = (__bf16*)(ws + OFF_H1B);
    __bf16* att_b  = (__bf16*)(ws + OFF_ATTB);
    float*  gi2    = (float*)(ws + OFF_GI2);
    float*  gh2    = (float*)(ws + OFF_GH2);
    float*  h2     = (float*)(ws + OFF_H2);
    __bf16* h2_b   = (__bf16*)(ws + OFF_H2B);
    float*  o      = gi2;  // alias: gi2 is dead by the time o is produced

    float* outp   = (float*)d_out;                    // [BB, HD]
    float* alphas = (float*)d_out + (size_t)BB * HD;  // [BB, TT, CD]

    const int TPB = 256;
    auto blocks = [](int n) { return (n + 255) / 256; };

    // One-time conversions to bf16
    cvt_f32_bf16<<<blocks(CD * VD),     TPB, 0, stream>>>(Wv,    wv_b,   CD * VD);
    cvt_f32_bf16<<<blocks(CD * QD),     TPB, 0, stream>>>(Wq,    wq_b,   CD * QD);
    cvt_f32_bf16<<<blocks(3 * CD * CD), TPB, 0, stream>>>(w_ih1, wih1_b, 3 * CD * CD);
    cvt_f32_bf16<<<blocks(3 * CD * CD), TPB, 0, stream>>>(w_hh1, whh1_b, 3 * CD * CD);
    cvt_f32_bf16<<<blocks(3 * HD * CD), TPB, 0, stream>>>(w_ih2, wih2_b, 3 * HD * CD);
    cvt_f32_bf16<<<blocks(3 * HD * HD), TPB, 0, stream>>>(w_hh2, whh2_b, 3 * HD * HD);
    cvt_f32_bf16<<<blocks(HD * HD),     TPB, 0, stream>>>(Wfc,   wfc_b,  HD * HD);
    cvt_f32_bf16<<<blocks(BB * VD),     TPB, 0, stream>>>(v,     v_b,    BB * VD);
    cvt_f32_bf16<<<blocks(BB * QD),     TPB, 0, stream>>>(q,     q_b,    BB * QD);

    // Init h1/h2 (+shadows), output maxpool accumulator
    init_state<<<blocks(BB * HD), TPB, 0, stream>>>(h1, h1_b, h2, h2_b, outp);

    // Prologue: fv = leaky(v @ Wv^T), fq = leaky(q @ Wq^T)
    gemm_bf16_wmma<<<dim3(CD / 128, BB / 128), 256, 0, stream>>>(v_b, VD, wv_b, fv, CD, VD, 1);
    gemm_bf16_wmma<<<dim3(CD / 128, BB / 128), 256, 0, stream>>>(q_b, QD, wq_b, fq, CD, QD, 1);

    for (int t = 0; t < TT; ++t) {
        // xt_b = bf16(caption[:, t, :])
        cvt_caption_slice<<<blocks(BB * CD), TPB, 0, stream>>>(caption, xt_b, t);
        // gi1 = x_t @ w_ih1^T
        gemm_bf16_wmma<<<dim3((3 * CD) / 128, BB / 128), 256, 0, stream>>>(
            xt_b, CD, wih1_b, gi1, 3 * CD, CD, 0);
        // gh1 = h1 @ w_hh1^T
        gemm_bf16_wmma<<<dim3((3 * CD) / 128, BB / 128), 256, 0, stream>>>(
            h1_b, CD, whh1_b, gh1, 3 * CD, CD, 0);
        // GRU1 gates + attention + alphas
        gru1_att_kernel<<<blocks(BB * CD), TPB, 0, stream>>>(
            gi1, gh1, b_ih1, b_hh1, h1, h1_b, fv, fq, caption, cap_len, att_b, alphas, t);
        // gi2 = att @ w_ih2^T
        gemm_bf16_wmma<<<dim3((3 * HD) / 128, BB / 128), 256, 0, stream>>>(
            att_b, CD, wih2_b, gi2, 3 * HD, CD, 0);
        // gh2 = h2 @ w_hh2^T
        gemm_bf16_wmma<<<dim3((3 * HD) / 128, BB / 128), 256, 0, stream>>>(
            h2_b, HD, whh2_b, gh2, 3 * HD, HD, 0);
        // GRU2 gates
        gru2_kernel<<<blocks(BB * HD), TPB, 0, stream>>>(
            gi2, gh2, b_ih2, b_hh2, h2, h2_b, cap_len, t);
        // o = h2 @ Wfc^T   (written into the now-dead gi2 region)
        gemm_bf16_wmma<<<dim3(HD / 128, BB / 128), 256, 0, stream>>>(
            h2_b, HD, wfc_b, o, HD, HD, 0);
        // out = max(out, active ? leaky(o) : 0)
        maxpool_kernel<<<blocks(BB * HD), TPB, 0, stream>>>(o, cap_len, outp, t);
    }
}
